// MLPSimDirectNormConstructor_26508538151435
// MI455X (gfx1250) — matrix operand order repaired and checked by reading, not compiled, not run on hardware
//
#include <hip/hip_runtime.h>
#include <hip/hip_bf16.h>

// ---------------- problem constants (from setup_inputs) ----------------
#define B_   4
#define N_   4096
#define T_   512
#define D_   32
#define NT_  4608            // N + T
#define ALPHA_ 3.0f
#define EPS_  1e-30f

typedef __attribute__((ext_vector_type(16))) _Float16 v16h;
typedef __attribute__((ext_vector_type(8)))  _Float16 v8h;
typedef __attribute__((ext_vector_type(8)))  float    v8f;
typedef __attribute__((ext_vector_type(4)))  float    v4f;

// ---------------- WMMA helper (wave32, 16x16x32 f16 -> f32) ----------------
__device__ __forceinline__ v8f wmma_f16(v16h a, v16h b, v8f c) {
    // 8 args: (neg_a, A, neg_b, B, c_mod, C, reuse_a, reuse_b)
    return __builtin_amdgcn_wmma_f32_16x16x32_f16(false, a, false, b, (short)0, c,
                                                  false, false);
}

// A-matrix 16x32 f16 fragment from row-major [rows x 32] source.
// lanes 0-15 -> M=lane, K={0..7,16..23}; lanes 16-31 -> M=lane-16, K={8..15,24..31}.
__device__ __forceinline__ v16h load_a_frag(const _Float16* base, int row0) {
    const int lane = threadIdx.x & 31;
    const int kb   = (lane < 16) ? 0 : 8;
    const _Float16* p = base + (row0 + (lane & 15)) * D_ + kb;
    v8h lo = *(const v8h*)(p);        // K = kb .. kb+7      (16B aligned)
    v8h hi = *(const v8h*)(p + 16);   // K = kb+16 .. kb+23  (16B aligned)
    v16h f;
#pragma unroll
    for (int i = 0; i < 8; ++i) { f[i] = lo[i]; f[i + 8] = hi[i]; }
    return f;
}

// B-matrix 32x16 f16 fragment, B = (rows at col0)^T of a row-major [cols x 32] source:
// lanes 0-15 -> N=lane, K={0..7,16..23}; lanes 16-31 -> N=lane-16, K={8..15,24..31}.
__device__ __forceinline__ v16h load_b_frag(const _Float16* base, int col0) {
    const int lane = threadIdx.x & 31;
    const _Float16* p = base + (col0 + (lane & 15)) * D_ + ((lane < 16) ? 0 : 8);
    v8h lo = *(const v8h*)(p);
    v8h hi = *(const v8h*)(p + 16);
    v16h f;
#pragma unroll
    for (int i = 0; i < 8; ++i) { f[i] = lo[i]; f[i + 8] = hi[i]; }
    return f;
}

// ---------------- kernel 0: zero the 4 relu-max slots ----------------
__global__ void init_max_kernel(unsigned* maxes) {
    if (threadIdx.x < 4) maxes[threadIdx.x] = 0u;
}

// ---------------- kernel 1: n1/n2 = f16(tanh(3 * x_s @ Wss^T)) ----------------
__global__ void proj_kernel(const float* __restrict__ xs,
                            const float* __restrict__ w1,
                            const float* __restrict__ w2,
                            _Float16* __restrict__ n1, _Float16* __restrict__ n2) {
    int idx = blockIdx.x * blockDim.x + threadIdx.x;   // over B*N*D
    if (idx >= B_ * N_ * D_) return;
    int e  = idx & (D_ - 1);
    int bn = idx >> 5;
    const float* x  = xs + bn * D_;
    const float* r1 = w1 + e * D_;   // einsum 'bnd,ed' -> W row e dotted with x
    const float* r2 = w2 + e * D_;
    float d1 = 0.f, d2 = 0.f;
#pragma unroll
    for (int d = 0; d < D_; ++d) { float xv = x[d]; d1 += xv * r1[d]; d2 += xv * r2[d]; }
    n1[idx] = (_Float16)tanhf(ALPHA_ * d1);
    n2[idx] = (_Float16)tanhf(ALPHA_ * d2);
}

// ---------------- kernel 2: spatial dot-vectors for st / ts ----------------
__global__ void spatial_vec_kernel(const float* __restrict__ xs,
                                   const float* __restrict__ st_w,
                                   const float* __restrict__ ts_w,
                                   float* __restrict__ a_st, float* __restrict__ c_ts) {
    int idx = blockIdx.x * blockDim.x + threadIdx.x;   // over B*N
    if (idx >= B_ * N_) return;
    const float* x = xs + idx * D_;
    float a = 0.f, c = 0.f;
#pragma unroll
    for (int d = 0; d < D_; ++d) { float xv = x[d]; a += xv * st_w[d]; c += xv * ts_w[D_ + d]; }
    a_st[idx] = a;   // x_s . st_w[:D]
    c_ts[idx] = c;   // x_s . ts_w[D:]
}

// ---------------- kernel 3: f16 copy of x_t + temporal dot-vectors ----------------
__global__ void temporal_prep_kernel(const float* __restrict__ xt,
                                     const float* __restrict__ st_w,
                                     const float* __restrict__ ts_w,
                                     _Float16* __restrict__ xt_h,
                                     float* __restrict__ a_ts, float* __restrict__ c_st) {
    int idx = blockIdx.x * blockDim.x + threadIdx.x;   // over B*T
    if (idx >= B_ * T_) return;
    const float* x = xt + idx * D_;
    float a = 0.f, c = 0.f;
#pragma unroll
    for (int d = 0; d < D_; ++d) {
        float xv = x[d];
        xt_h[idx * D_ + d] = (_Float16)xv;
        a += xv * ts_w[d];        // x_t . ts_w[:D]
        c += xv * st_w[D_ + d];   // x_t . st_w[D:]
    }
    a_ts[idx] = a;
    c_st[idx] = c;
}

// ---------------- kernel 4: triangular WMMA GEMM, one wave per 16x64 strip ----
// ANTISYM=true  (ss): tile = scale*(A@B^T - B@A^T); compute upper triangle only,
//                     mirror-store -tile^T into the lower triangle (exact by
//                     antisymmetry); relu-max == max|v|.
// ANTISYM=false (tt): tile = scale*(A@A^T); symmetric + later triu-masked, so
//                     lower-triangle tiles are skipped entirely (norm pass zeros
//                     them); relu-max attained in the computed upper triangle.
template <bool ANTISYM>
__global__ void wmma_tri_kernel(const _Float16* __restrict__ Abase,
                                const _Float16* __restrict__ Bbase,
                                int rows, int stripsPerRow, int row_off, int col_off,
                                float scale, float* __restrict__ out,
                                unsigned* __restrict__ maxslot) {
    const int waveId = threadIdx.x >> 5;
    const int lane   = threadIdx.x & 31;
    const int tilesPerDim    = rows >> 4;
    const int stripsPerBatch = tilesPerDim * stripsPerRow;
    const int s = blockIdx.x * (blockDim.x >> 5) + waveId;

    float lmax = 0.f;
    if (s < B_ * stripsPerBatch) {
        const int b    = s / stripsPerBatch;
        const int rem  = s - b * stripsPerBatch;
        const int trow = rem / stripsPerRow;
        const int tc0  = (rem - trow * stripsPerRow) * 4;

        if (tc0 + 3 >= trow) {                      // strip touches upper triangle
            const _Float16* A  = Abase + (size_t)b * rows * D_;
            const _Float16* Bm = Bbase + (size_t)b * rows * D_;

            v16h a1 = load_a_frag(A, trow * 16);    // reused across 4 col tiles
            v16h a2;
            if (ANTISYM) a2 = load_a_frag(Bm, trow * 16);

            const int hi8  = (lane < 16) ? 0 : 8;
            const int ln   = lane & 15;
            const int rowg = row_off + trow * 16 + hi8;   // first direct row

#pragma unroll
            for (int j = 0; j < 4; ++j) {
                const int tcol = tc0 + j;
                if (tcol < trow) continue;          // below diagonal: skip (wave-uniform)

                v8f zero = {};
                v16h b1 = load_b_frag(Bm, tcol * 16);
                v8f acc = wmma_f16(a1, b1, zero);   // v_wmma_f32_16x16x32_f16
                if (ANTISYM) {
                    v16h b2 = load_b_frag(A, tcol * 16);
                    v8f acc2 = wmma_f16(a2, b2, zero);
                    acc = acc - acc2;
                }

                const int colg = col_off + tcol * 16 + ln;
                float vv[8];
#pragma unroll
                for (int r = 0; r < 8; ++r) {
                    float v = acc[r] * scale;
                    vv[r] = v;
                    out[((size_t)b * NT_ + (rowg + r)) * NT_ + colg] = v;
                    lmax = fmaxf(lmax, ANTISYM ? fabsf(v) : v);
                }

                if (ANTISYM && tcol != trow) {
                    // mirror: adj[mrow][mcol..mcol+7] = -vv[0..7], contiguous per lane
                    const int mrow = row_off + tcol * 16 + ln;
                    const int mcol = col_off + trow * 16 + hi8;
                    float* p = out + ((size_t)b * NT_ + mrow) * NT_ + mcol;
                    v4f m0 = { -vv[0], -vv[1], -vv[2], -vv[3] };
                    v4f m1 = { -vv[4], -vv[5], -vv[6], -vv[7] };
                    *(v4f*)(p)     = m0;            // 32B-aligned (mcol % 8 == 0)
                    *(v4f*)(p + 4) = m1;
                }
            }
        }
    }

    __shared__ unsigned smax;
    if (threadIdx.x == 0) smax = 0u;
    __syncthreads();
    atomicMax(&smax, __float_as_uint(lmax));        // values >= 0: bits are monotone
    __syncthreads();
    if (threadIdx.x == 0) atomicMax(maxslot, smax);
}

// ---------------- kernel 5a: st rank-1 fill (row-major, float4) ----------------
__global__ void st_fill_kernel(const float* __restrict__ a_st, const float* __restrict__ c_st,
                               const float* __restrict__ st_b,
                               float* __restrict__ out, unsigned* __restrict__ maxslot) {
    int idx = blockIdx.x * blockDim.x + threadIdx.x;   // over B*N*(T/4)
    float lmax = 0.f;
    if (idx < B_ * N_ * (T_ / 4)) {
        int q = idx % (T_ / 4);
        int n = (idx / (T_ / 4)) % N_;
        int b = idx / ((T_ / 4) * N_);
        float a = a_st[b * N_ + n] + st_b[0];
        v4f c = *(const v4f*)(c_st + b * T_ + q * 4);
        v4f v = { a + c.x, a + c.y, a + c.z, a + c.w };
        *(v4f*)(out + ((size_t)b * NT_ + n) * NT_ + N_ + q * 4) = v;
        lmax = fmaxf(fmaxf(fmaxf(v.x, v.y), fmaxf(v.z, v.w)), 0.f);
    }
    __shared__ unsigned smax;
    if (threadIdx.x == 0) smax = 0u;
    __syncthreads();
    atomicMax(&smax, __float_as_uint(lmax));
    __syncthreads();
    if (threadIdx.x == 0) atomicMax(maxslot, smax);
}

// ---------------- kernel 5b: ts rank-1 fill (row-major, float4) ----------------
__global__ void ts_fill_kernel(const float* __restrict__ a_ts, const float* __restrict__ c_ts,
                               const float* __restrict__ ts_b,
                               float* __restrict__ out, unsigned* __restrict__ maxslot) {
    int idx = blockIdx.x * blockDim.x + threadIdx.x;   // over B*T*(N/4)
    float lmax = 0.f;
    if (idx < B_ * T_ * (N_ / 4)) {
        int q = idx % (N_ / 4);
        int t = (idx / (N_ / 4)) % T_;
        int b = idx / ((N_ / 4) * T_);
        float a = a_ts[b * T_ + t] + ts_b[0];
        v4f c = *(const v4f*)(c_ts + b * N_ + q * 4);
        v4f v = { a + c.x, a + c.y, a + c.z, a + c.w };
        *(v4f*)(out + ((size_t)b * NT_ + N_ + t) * NT_ + q * 4) = v;
        lmax = fmaxf(fmaxf(fmaxf(v.x, v.y), fmaxf(v.z, v.w)), 0.f);
    }
    __shared__ unsigned smax;
    if (threadIdx.x == 0) smax = 0u;
    __syncthreads();
    atomicMax(&smax, __float_as_uint(lmax));
    __syncthreads();
    if (threadIdx.x == 0) atomicMax(maxslot, smax);
}

// ---------------- kernel 6: uni_adj normalize + triu(tt), in place, float4 ----
__global__ void norm_kernel(float* __restrict__ out, const unsigned* __restrict__ maxes) {
    const float m_ss = __uint_as_float(maxes[0]) + EPS_;
    const float m_st = __uint_as_float(maxes[1]) + EPS_;
    const float m_ts = __uint_as_float(maxes[2]) + EPS_;
    const float m_tt = __uint_as_float(maxes[3]) + EPS_;
    const size_t total4 = (size_t)B_ * NT_ * NT_ / 4;
    const size_t stride = (size_t)gridDim.x * blockDim.x;
    for (size_t i4 = (size_t)blockIdx.x * blockDim.x + threadIdx.x; i4 < total4; i4 += stride) {
        const size_t e  = i4 * 4;
        const size_t rc = e % ((size_t)NT_ * NT_);
        const int r  = (int)(rc / NT_);
        const int c0 = (int)(rc % NT_);      // region boundary (4096) is 4-aligned
        float denom;
        bool tt = false;
        if (r < N_)       denom = (c0 < N_) ? m_ss : m_st;
        else if (c0 < N_) denom = m_ts;
        else { denom = m_tt; tt = true; }
        v4f v = *(v4f*)(out + e);
        v4f o;
#pragma unroll
        for (int k = 0; k < 4; ++k) {
            float t = tanhf(fmaxf(v[k], 0.f) / denom);
            if (t != t) t = 0.f;                           // nan_to_num
            o[k] = (tt && (c0 + k) < r) ? 0.f : t;         // triu keeps col >= row
        }
        *(v4f*)(out + e) = o;
    }
}

// ---------------- host: launch sequence ----------------
extern "C" void kernel_launch(void* const* d_in, const int* in_sizes, int n_in,
                              void* d_out, int out_size, void* d_ws, size_t ws_size,
                              hipStream_t stream) {
    const float* xs   = (const float*)d_in[0];   // [B,N,D]
    const float* xt   = (const float*)d_in[1];   // [B,T,D]
    const float* ss1w = (const float*)d_in[2];   // [D,D]
    const float* ss2w = (const float*)d_in[3];   // [D,D]
    const float* st_w = (const float*)d_in[4];   // [1,2D]
    const float* st_b = (const float*)d_in[5];   // [1]
    const float* ts_w = (const float*)d_in[6];   // [1,2D]
    const float* ts_b = (const float*)d_in[7];   // [1]
    float* out = (float*)d_out;

    // workspace layout (~2.4 MB, all 16B-aligned)
    _Float16* n1   = (_Float16*)d_ws;                       // B*N*D halves
    _Float16* n2   = n1 + (size_t)B_ * N_ * D_;             // B*N*D halves
    _Float16* xt_h = n2 + (size_t)B_ * N_ * D_;             // B*T*D halves
    float*    a_st = (float*)(xt_h + (size_t)B_ * T_ * D_); // B*N
    float*    c_ts = a_st + B_ * N_;                        // B*N
    float*    a_ts = c_ts + B_ * N_;                        // B*T
    float*    c_st = a_ts + B_ * T_;                        // B*T
    unsigned* maxes = (unsigned*)(c_st + B_ * T_);          // 4 slots

    init_max_kernel<<<1, 32, 0, stream>>>(maxes);

    proj_kernel<<<(B_ * N_ * D_) / 256, 256, 0, stream>>>(xs, ss1w, ss2w, n1, n2);
    spatial_vec_kernel<<<(B_ * N_ + 255) / 256, 256, 0, stream>>>(xs, st_w, ts_w, a_st, c_ts);
    temporal_prep_kernel<<<(B_ * T_ + 255) / 256, 256, 0, stream>>>(xt, st_w, ts_w, xt_h, a_ts, c_st);

    // ss: antisymmetric WMMA GEMM on upper-triangle strips of 4 tiles, 8 waves/block
    const int ssStrips = B_ * (N_ / 16) * (N_ / 64);        // 4 * 256 * 64 = 65536
    wmma_tri_kernel<true><<<ssStrips / 8, 256, 0, stream>>>(
        n1, n2, N_, N_ / 64, 0, 0, ALPHA_, out, &maxes[0]);

    // tt: x_t @ x_t^T, symmetric + triu-masked -> upper-triangle strips only
    const int ttStrips = B_ * (T_ / 16) * (T_ / 64);        // 4 * 32 * 8 = 1024
    wmma_tri_kernel<false><<<ttStrips / 8, 256, 0, stream>>>(
        xt_h, xt_h, T_, T_ / 64, N_, N_, 1.0f, out, &maxes[3]);

    st_fill_kernel<<<(B_ * N_ * (T_ / 4)) / 256, 256, 0, stream>>>(
        a_st, c_st, st_b, out, &maxes[1]);
    ts_fill_kernel<<<(B_ * T_ * (N_ / 4)) / 256, 256, 0, stream>>>(
        a_ts, c_ts, ts_b, out, &maxes[2]);

    norm_kernel<<<4096, 256, 0, stream>>>(out, maxes);
}
